// GAT_L2_63831803953269
// MI455X (gfx1250) — matrix operand order repaired and checked by reading, not compiled, not run on hardware
//
#include <hip/hip_runtime.h>
#include <hip/hip_bf16.h>
#include <math.h>

// ---------------- constants from the reference ----------------
#define H1 4
#define C1_ 32
#define F1 128          // H1*C1
#define C2_ 16
#define NEG_SLOPE 0.2f
#define EPS_ 1e-16f

typedef __attribute__((ext_vector_type(16))) __bf16 v16bf;
typedef __attribute__((ext_vector_type(8)))  float  v8f;

#define LDS_STRIDE 36   // padded row stride (words): %4==0 keeps float4 align, spreads 64 banks

// =====================================================================
// Split-precision bf16x3 GEMM:  D[M,N] = A[M,K] * B[K,N]  (fp32 in/out)
// One wave computes a 16 x (16*NT) output strip; K stepped by 32 through
// padded LDS staging (A row-major, B transposed to [col][k] so both
// operands gather as float4). 3 WMMAs per 16x16 tile per K-step:
//   x*w ~= xl*wh + xh*wl + xh*wh   (fp32 accumulate, ~fp32 accuracy)
// Requires M%16==0, K%32==0, N%(16*NT)==0 (true here).
// =====================================================================
template <int NT>
__global__ __launch_bounds__(32) void gat_gemm_bf16x3(
    const float* __restrict__ A, const float* __restrict__ B,
    float* __restrict__ D, int M, int K, int N)
{
    __shared__ float sA[16 * LDS_STRIDE];          // [m][k]
    __shared__ float sBt[NT * 16 * LDS_STRIDE];    // [col][k] (transposed)

    const int lane = threadIdx.x;                  // 0..31 (wave32)
    const int m0 = blockIdx.y << 4;
    const int n0 = blockIdx.x * (16 * NT);

    const int mrow = lane & 15;                    // M row (A) / N col (B) within tile
    const int hi16 = lane >> 4;                    // 0 or 1
    const int kbA  = hi16 * 8;                     // A: lanes16-31 hold K=8..15,24..31
    const int kbB  = hi16 * 8;                     // B: lanes16-31 hold K=8..15,24..31

    v8f acc[NT];
#pragma unroll
    for (int t = 0; t < NT; ++t) acc[t] = (v8f){};

    for (int k0 = 0; k0 < K; k0 += 32) {
        // ---- cooperative, coalesced staging ----
        for (int j = lane; j < 512; j += 32)       // A tile: 16x32
            sA[(j >> 5) * LDS_STRIDE + (j & 31)] =
                A[(size_t)(m0 + (j >> 5)) * K + (k0 + (j & 31))];
        for (int j = lane; j < NT * 512; j += 32) {// B tile: 32 x (16*NT), store transposed
            const int col = j % (NT * 16);
            const int kk  = j / (NT * 16);
            sBt[col * LDS_STRIDE + kk] = B[(size_t)(k0 + kk) * N + (n0 + col)];
        }
        __syncthreads();

        // ---- A operand: 4x float4 LDS loads, then split hi/lo bf16 ----
        float af[16];
        {
            const float* pa = &sA[mrow * LDS_STRIDE + kbA];
            *(float4*)&af[0]  = *(const float4*)&pa[0];
            *(float4*)&af[4]  = *(const float4*)&pa[4];
            *(float4*)&af[8]  = *(const float4*)&pa[16];   // i>=8 -> K = kbA+16..kbA+23
            *(float4*)&af[12] = *(const float4*)&pa[20];
        }
        v16bf ah, al;
#pragma unroll
        for (int i = 0; i < 16; ++i) {
            const __bf16 xh = (__bf16)af[i];
            ah[i] = xh;
            al[i] = (__bf16)(af[i] - (float)xh);
        }

        // ---- per column-tile: B operand + 3 WMMAs ----
#pragma unroll
        for (int t = 0; t < NT; ++t) {
            float bf[16];
            const float* pb = &sBt[(t * 16 + mrow) * LDS_STRIDE + kbB];
            *(float4*)&bf[0]  = *(const float4*)&pb[0];    // K = kbB+0..7, kbB+16..23 (table order)
            *(float4*)&bf[4]  = *(const float4*)&pb[4];
            *(float4*)&bf[8]  = *(const float4*)&pb[16];
            *(float4*)&bf[12] = *(const float4*)&pb[20];
            v16bf bh, bl;
#pragma unroll
            for (int i = 0; i < 16; ++i) {
                const __bf16 wh = (__bf16)bf[i];
                bh[i] = wh;
                bl[i] = (__bf16)(bf[i] - (float)wh);
            }
            acc[t] = __builtin_amdgcn_wmma_f32_16x16x32_bf16(false, al, false, bh, (short)0, acc[t], false, false);
            acc[t] = __builtin_amdgcn_wmma_f32_16x16x32_bf16(false, ah, false, bl, (short)0, acc[t], false, false);
            acc[t] = __builtin_amdgcn_wmma_f32_16x16x32_bf16(false, ah, false, bh, (short)0, acc[t], false, false);
        }
        __syncthreads();
    }

    // C/D layout: lane<16 -> N=lane, M=r; lane>=16 -> N=lane-16, M=r+8
    const int mbase = m0 + hi16 * 8;
#pragma unroll
    for (int t = 0; t < NT; ++t) {
        const int ncol = n0 + t * 16 + mrow;
#pragma unroll
        for (int r = 0; r < 8; ++r)
            D[(size_t)(mbase + r) * N + ncol] = acc[t][r];
    }
}

// =====================================================================
// Per-node attention logits: as[n,h] = <h[n,h,:], a_src[h,:]>, same for ad
// =====================================================================
__global__ void gat_alpha_kernel(const float* __restrict__ h,
                                 const float* __restrict__ a_src,
                                 const float* __restrict__ a_dst,
                                 float* __restrict__ as, float* __restrict__ ad,
                                 int N, int H, int C)
{
    const int t = blockIdx.x * blockDim.x + threadIdx.x;
    if (t >= N * H) return;
    const int n = t / H, hh = t % H;
    const float* hp  = h + (size_t)n * H * C + hh * C;
    const float* asp = a_src + hh * C;
    const float* adp = a_dst + hh * C;
    float s1 = 0.f, s2 = 0.f;
    for (int c = 0; c < C; ++c) { const float v = hp[c]; s1 += v * asp[c]; s2 += v * adp[c]; }
    as[t] = s1; ad[t] = s2;
}

// ---------------- helpers ----------------
__device__ __forceinline__ void edge_sd(const int* __restrict__ ei, int E, int e,
                                        int& s, int& d)
{
    if (e < E) { s = ei[e]; d = ei[E + e]; }
    else       { s = d = e - E; }            // appended self-loops
}

__device__ __forceinline__ float leaky(float v) { return v > 0.f ? v : NEG_SLOPE * v; }

// order-preserving float max via signed-max / unsigned-min split
__device__ __forceinline__ void atomicMaxF(float* addr, float val)
{
    const int iv = __float_as_int(val);
    if (iv >= 0) atomicMax((int*)addr, iv);
    else         atomicMin((unsigned int*)addr, (unsigned int)iv);
}

// =====================================================================
// Pass 1: segment max over incoming edges
// =====================================================================
__global__ void gat_edge_max_kernel(const int* __restrict__ ei, int E, int Etot, int H,
                                    const float* __restrict__ as,
                                    const float* __restrict__ ad,
                                    float* __restrict__ m)
{
    const int t = blockIdx.x * blockDim.x + threadIdx.x;
    if (t >= Etot * H) return;
    const int e = t / H, hh = t % H;
    int s, d; edge_sd(ei, E, e, s, d);
    const float v = leaky(as[s * H + hh] + ad[d * H + hh]);
    atomicMaxF(&m[d * H + hh], v);
}

// =====================================================================
// Pass 2: segment sum of exp(e - m[dst])
// =====================================================================
__global__ void gat_edge_sum_kernel(const int* __restrict__ ei, int E, int Etot, int H,
                                    const float* __restrict__ as,
                                    const float* __restrict__ ad,
                                    const float* __restrict__ m,
                                    float* __restrict__ ssum)
{
    const int t = blockIdx.x * blockDim.x + threadIdx.x;
    if (t >= Etot * H) return;
    const int e = t / H, hh = t % H;
    int s, d; edge_sd(ei, E, e, s, d);
    const float v = leaky(as[s * H + hh] + ad[d * H + hh]);
    atomicAdd(&ssum[d * H + hh], expf(v - m[d * H + hh]));
}

// =====================================================================
// Pass 3: weighted aggregation. One wave per edge; lanes<H compute alpha,
// broadcast by shuffle, all lanes scatter features with atomicAdd.
// =====================================================================
__global__ void gat_agg_kernel(const int* __restrict__ ei, int E, int Etot, int H, int C,
                               const float* __restrict__ as,
                               const float* __restrict__ ad,
                               const float* __restrict__ m,
                               const float* __restrict__ ssum,
                               const float* __restrict__ hbuf,
                               float* __restrict__ out)
{
    const int gid  = blockIdx.x * blockDim.x + threadIdx.x;
    const int e    = gid >> 5;            // wave-uniform
    const int lane = threadIdx.x & 31;
    if (e >= Etot) return;
    int s, d; edge_sd(ei, E, e, s, d);

    float alpha = 0.f;
    if (lane < H) {
        const float v  = leaky(as[s * H + lane] + ad[d * H + lane]);
        const float ex = expf(v - m[d * H + lane]);
        alpha = ex / (ssum[d * H + lane] + EPS_);
    }
    const int F = H * C;
    for (int f = lane; f < F; f += 32) {
        const float a = __shfl(alpha, f / C, 32);
        atomicAdd(&out[(size_t)d * F + f], hbuf[(size_t)s * F + f] * a);
    }
}

// =====================================================================
// Init: zero accumulators / outputs, -inf the segment maxima
// =====================================================================
__global__ void gat_init_kernel(float* out1, float* m1, float* sum1,
                                float* m2, float* sum2, float* dout, int N)
{
    const int t = blockIdx.x * blockDim.x + threadIdx.x;
    if (t < N * F1)  out1[t] = 0.f;
    if (t < N * H1)  { m1[t] = -3.0e38f; sum1[t] = 0.f; }
    if (t < N)       { m2[t] = -3.0e38f; sum2[t] = 0.f; }
    if (t < N * C2_) dout[t] = 0.f;
}

// h2in = elu(out1 + b1)
__global__ void gat_bias_elu_kernel(const float* __restrict__ out1,
                                    const float* __restrict__ b1,
                                    float* __restrict__ h2in, int N)
{
    const int t = blockIdx.x * blockDim.x + threadIdx.x;
    if (t >= N * F1) return;
    const float v = out1[t] + b1[t % F1];
    h2in[t] = v > 0.f ? v : (expf(v) - 1.f);
}

// d_out += b2
__global__ void gat_final_bias_kernel(float* __restrict__ dout,
                                      const float* __restrict__ b2, int total)
{
    const int t = blockIdx.x * blockDim.x + threadIdx.x;
    if (t >= total) return;
    dout[t] += b2[t % C2_];
}

// =====================================================================
extern "C" void kernel_launch(void* const* d_in, const int* in_sizes, int n_in,
                              void* d_out, int out_size, void* d_ws, size_t ws_size,
                              hipStream_t stream)
{
    const float* x      = (const float*)d_in[0];
    const int*   ei     = (const int*)  d_in[1];
    const float* W1     = (const float*)d_in[2];
    const float* a_src1 = (const float*)d_in[3];
    const float* a_dst1 = (const float*)d_in[4];
    const float* b1     = (const float*)d_in[5];
    const float* W2     = (const float*)d_in[6];
    const float* a_src2 = (const float*)d_in[7];
    const float* a_dst2 = (const float*)d_in[8];
    const float* b2     = (const float*)d_in[9];
    float* out = (float*)d_out;

    const int Fin  = in_sizes[2] / in_sizes[3];   // (Fin*H1*C1)/(H1*C1) = 256
    const int N    = in_sizes[0] / Fin;           // 50000
    const int E    = in_sizes[1] / 2;             // 800000
    const int Etot = E + N;                       // + self loops

    // ---- workspace carve-out (floats) ----
    float* ws = (float*)d_ws;
    size_t o = 0;
    float* h1   = ws + o; o += (size_t)N * F1;    // layer-1 features (reused as elu(h) input to GEMM2)
    float* out1 = ws + o; o += (size_t)N * F1;    // layer-1 aggregation
    float* as1  = ws + o; o += (size_t)N * H1;
    float* ad1  = ws + o; o += (size_t)N * H1;
    float* m1   = ws + o; o += (size_t)N * H1;
    float* sum1 = ws + o; o += (size_t)N * H1;
    float* h2   = ws + o; o += (size_t)N * C2_;   // layer-2 features
    float* as2  = ws + o; o += (size_t)N;
    float* ad2  = ws + o; o += (size_t)N;
    float* m2   = ws + o; o += (size_t)N;
    float* sum2 = ws + o; o += (size_t)N;
    (void)ws_size; (void)n_in; (void)out_size;

    const int BT = 256;
    auto blocks = [](long long n, int b) { return (unsigned)((n + b - 1) / b); };

    // 0) init accumulators (re-run every call => deterministic replay)
    gat_init_kernel<<<blocks((long long)N * F1, BT), BT, 0, stream>>>(out1, m1, sum1, m2, sum2, out, N);

    // ---------------- layer 1 ----------------
    {
        dim3 g(F1 / 64, N / 16);                  // NT=4 -> 64 cols per wave
        gat_gemm_bf16x3<4><<<g, 32, 0, stream>>>(x, W1, h1, N, Fin, F1);
    }
    gat_alpha_kernel<<<blocks((long long)N * H1, BT), BT, 0, stream>>>(h1, a_src1, a_dst1, as1, ad1, N, H1, C1_);
    gat_edge_max_kernel<<<blocks((long long)Etot * H1, BT), BT, 0, stream>>>(ei, E, Etot, H1, as1, ad1, m1);
    gat_edge_sum_kernel<<<blocks((long long)Etot * H1, BT), BT, 0, stream>>>(ei, E, Etot, H1, as1, ad1, m1, sum1);
    gat_agg_kernel<<<blocks((long long)Etot * 32, BT), BT, 0, stream>>>(ei, E, Etot, H1, C1_, as1, ad1, m1, sum1, h1, out1);
    gat_bias_elu_kernel<<<blocks((long long)N * F1, BT), BT, 0, stream>>>(out1, b1, h1, N);  // h1 <- elu(out1+b1)

    // ---------------- layer 2 ----------------
    {
        dim3 g(C2_ / 16, N / 16);                 // NT=1 -> 16 cols per wave
        gat_gemm_bf16x3<1><<<g, 32, 0, stream>>>(h1, W2, h2, N, F1, C2_);
    }
    gat_alpha_kernel<<<blocks((long long)N, BT), BT, 0, stream>>>(h2, a_src2, a_dst2, as2, ad2, N, 1, C2_);
    gat_edge_max_kernel<<<blocks((long long)Etot, BT), BT, 0, stream>>>(ei, E, Etot, 1, as2, ad2, m2);
    gat_edge_sum_kernel<<<blocks((long long)Etot, BT), BT, 0, stream>>>(ei, E, Etot, 1, as2, ad2, m2, sum2);
    gat_agg_kernel<<<blocks((long long)Etot * 32, BT), BT, 0, stream>>>(ei, E, Etot, 1, C2_, as2, ad2, m2, sum2, h2, out);
    gat_final_bias_kernel<<<blocks((long long)N * C2_, BT), BT, 0, stream>>>(out, b2, N * C2_);
}
